// MultiKMeansLabeller_8796093022275
// MI455X (gfx1250) — matrix operand order repaired and checked by reading, not compiled, not run on hardware
//
#include <hip/hip_runtime.h>

// MI455X / gfx1250 wave32 WMMA k-means labeller.
// Computes argmin_k ( ||c1_k||^2 - 2 * x[:,64:128] . c1_k ) which equals the
// reference's returned labels (only the last slice's labels are returned).

typedef __attribute__((ext_vector_type(16))) _Float16 v16h;
typedef __attribute__((ext_vector_type(8)))  float    v8f;

#define N_PTS       131072
#define DIMS        128
#define SLICE_OFF   64
#define K_DIM       64
#define N_CENT      1024
#define CHUNK       256     // centroids staged per LDS pass (4 passes)
#define LDS_STRIDE  72      // halves per centroid row (64 + 8 pad -> bank spread)
#define ROWS_PER_WG 128     // 8 waves x 16-row M tiles
#define THREADS     256

__global__ __launch_bounds__(THREADS)
void kmeans_label_wmma(const float* __restrict__ inpt,
                       const float* __restrict__ centers1,
                       float* __restrict__ out)
{
    __shared__ _Float16 sC[CHUNK * LDS_STRIDE]; // 36 KB f16 centroid chunk
    __shared__ float    sN[CHUNK];              //  1 KB ||c||^2 chunk

    const int tid  = threadIdx.x;
    const int wave = tid >> 5;
    const int lane = tid & 31;
    const int l16  = lane & 15;
    const int h    = lane >> 4;          // half-wave select

    const int rowbase = blockIdx.x * ROWS_PER_WG + wave * 16;
    const int myrow   = rowbase + l16;   // A-matrix: M = lane % 16

    // ---- Load A fragments once: x slice dims [64,128) of this row, f32->f16.
    // 16-bit A 16x32 layout (contiguous, matching B): lane half i -> k = 16h+i.
    // A2 is the same shifted by K=32.
    const float4* xr4 = reinterpret_cast<const float4*>(
        inpt + (size_t)myrow * DIMS + SLICE_OFF);
    float4 c0 = xr4[4 * h + 0];          // k = 16h .. 16h+3
    float4 c1 = xr4[4 * h + 1];
    float4 c2 = xr4[4 * h + 2];          // k = 16h+8 ..
    float4 c3 = xr4[4 * h + 3];
    float4 c4 = xr4[8 + 4 * h + 0];      // k = 32+16h ..
    float4 c5 = xr4[8 + 4 * h + 1];
    float4 c6 = xr4[8 + 4 * h + 2];      // k = 40+16h ..
    float4 c7 = xr4[8 + 4 * h + 3];

    v16h a1, a2;
    a1[0]=(_Float16)c0.x;  a1[1]=(_Float16)c0.y;  a1[2]=(_Float16)c0.z;  a1[3]=(_Float16)c0.w;
    a1[4]=(_Float16)c1.x;  a1[5]=(_Float16)c1.y;  a1[6]=(_Float16)c1.z;  a1[7]=(_Float16)c1.w;
    a1[8]=(_Float16)c2.x;  a1[9]=(_Float16)c2.y;  a1[10]=(_Float16)c2.z; a1[11]=(_Float16)c2.w;
    a1[12]=(_Float16)c3.x; a1[13]=(_Float16)c3.y; a1[14]=(_Float16)c3.z; a1[15]=(_Float16)c3.w;
    a2[0]=(_Float16)c4.x;  a2[1]=(_Float16)c4.y;  a2[2]=(_Float16)c4.z;  a2[3]=(_Float16)c4.w;
    a2[4]=(_Float16)c5.x;  a2[5]=(_Float16)c5.y;  a2[6]=(_Float16)c5.z;  a2[7]=(_Float16)c5.w;
    a2[8]=(_Float16)c6.x;  a2[9]=(_Float16)c6.y;  a2[10]=(_Float16)c6.z; a2[11]=(_Float16)c6.w;
    a2[12]=(_Float16)c7.x; a2[13]=(_Float16)c7.y; a2[14]=(_Float16)c7.z; a2[15]=(_Float16)c7.w;

    float bestv[8];
    int   besti[8];
#pragma unroll
    for (int r = 0; r < 8; ++r) { bestv[r] = 3.4e38f; besti[r] = 0; }

    for (int cb = 0; cb < N_CENT; cb += CHUNK) {
        __syncthreads();   // all waves done reading previous chunk

        // ---- Stage chunk: one centroid row per thread, f32->f16 + norm.
        {
            const float4* crow4 = reinterpret_cast<const float4*>(
                centers1 + (size_t)(cb + tid) * K_DIM);
            _Float16* drow = &sC[tid * LDS_STRIDE];
            float nrm = 0.0f;
#pragma unroll
            for (int j = 0; j < 16; ++j) {
                float4 v = crow4[j];
                nrm += v.x * v.x + v.y * v.y + v.z * v.z + v.w * v.w;
                drow[4 * j + 0] = (_Float16)v.x;
                drow[4 * j + 1] = (_Float16)v.y;
                drow[4 * j + 2] = (_Float16)v.z;
                drow[4 * j + 3] = (_Float16)v.w;
            }
            sN[tid] = nrm;
        }
        // Warm next chunk into cache while we compute (gfx1250 global_prefetch).
        if (cb + CHUNK < N_CENT)
            __builtin_prefetch(centers1 + (size_t)(cb + CHUNK + tid) * K_DIM, 0, 0);
        __syncthreads();

        // ---- 16 N-tiles of 16 centroids each, 2x WMMA (K=32+32) per tile.
#pragma unroll 2
        for (int nt = 0; nt < CHUNK / 16; ++nt) {
            const int nl = nt * 16 + l16;     // B-matrix: N = lane % 16
            // 16-bit B 32x16 layout: lane half i -> k = 16h + i (contiguous).
            v16h b1 = *reinterpret_cast<const v16h*>(&sC[nl * LDS_STRIDE + 16 * h]);
            v16h b2 = *reinterpret_cast<const v16h*>(&sC[nl * LDS_STRIDE + 32 + 16 * h]);
            float cn = sN[nl];

            v8f acc = {};
            acc = __builtin_amdgcn_wmma_f32_16x16x32_f16(
                false, a1, false, b1, (short)0, acc, false, false);
            acc = __builtin_amdgcn_wmma_f32_16x16x32_f16(
                false, a2, false, b2, (short)0, acc, false, false);

            const int n = cb + nl;
#pragma unroll
            for (int r = 0; r < 8; ++r) {
                float d2 = cn - 2.0f * acc[r];
                if (d2 < bestv[r]) { bestv[r] = d2; besti[r] = n; }
            }
        }
    }

    // ---- Reduce argmin across the 16 lanes holding each row's 16 column slots.
    // C layout: VGPR r -> row r (lanes 0-15) and row r+8 (lanes 16-31).
#pragma unroll
    for (int r = 0; r < 8; ++r) {
#pragma unroll
        for (int m = 8; m >= 1; m >>= 1) {
            float ov = __shfl_xor(bestv[r], m, 16);
            int   oi = __shfl_xor(besti[r], m, 16);
            if (ov < bestv[r] || (ov == bestv[r] && oi < besti[r])) {
                bestv[r] = ov;
                besti[r] = oi;
            }
        }
        if (l16 == 0)
            out[rowbase + r + 8 * h] = (float)besti[r];
    }
}

extern "C" void kernel_launch(void* const* d_in, const int* in_sizes, int n_in,
                              void* d_out, int out_size, void* d_ws, size_t ws_size,
                              hipStream_t stream) {
    (void)in_sizes; (void)n_in; (void)out_size; (void)d_ws; (void)ws_size;
    const float* inpt     = (const float*)d_in[0];
    // d_in[1] = centers0 is unused: reference returns only the last slice's labels.
    const float* centers1 = (const float*)d_in[2];
    float*       out      = (float*)d_out;

    kmeans_label_wmma<<<dim3(N_PTS / ROWS_PER_WG), dim3(THREADS), 0, stream>>>(
        inpt, centers1, out);
}
